// SpikeFormerBlock_86397562126873
// MI455X (gfx1250) — matrix operand order repaired and checked by reading, not compiled, not run on hardware
//
#include <hip/hip_runtime.h>
#include <hip/hip_bf16.h>

// ---------------------------------------------------------------------------
// SpikeFormer block for MI455X (gfx1250).
// - all matmuls via v_wmma_f32_16x16x32_f16 (wave32, f32 accum)
// - weights pre-transposed to f16 so B-fragments load as global_load_b128
// - GEMM waves compute 32x64 C blocks (8 WMMAs share 2 A + 4 B fragments)
// - MLP GEMM stages its 64KB weight strip into LDS via the Tensor Data Mover
//   (tensor_load_to_lds + s_wait_tensorcnt) when the builtin is available
// - temporal LIF recurrences fused into producers (register / LDS membranes)
// ---------------------------------------------------------------------------

typedef __attribute__((ext_vector_type(16))) _Float16 v16h;
typedef __attribute__((ext_vector_type(2)))  _Float16 v2h;
typedef __attribute__((ext_vector_type(8)))  float    v8f;

#define BETA 0.951229424500714f   // exp(-1/20)
#define B_   8
#define T_   4
#define N_   256
#define D_   512
#define HID_ 2048
#define H_   8
#define DH_  64
#define M_   (B_ * T_ * N_)       // 8192 token rows

// ---- WMMA fragment helpers (layouts per CDNA5 ISA 7.12.2) ------------------

// A fragment: 16x32 f16 tile, row-major source (two contiguous 16B chunks/lane).
__device__ __forceinline__ v16h load_frag_a(const _Float16* __restrict__ A,
                                            int lda, int lane) {
  int m  = lane & 15;
  int kh = (lane >> 4) << 3;          // 0 or 8
  const _Float16* row = A + (size_t)m * lda;
  v16h a;
#pragma unroll
  for (int j = 0; j < 8; ++j) {
    int k = ((j >> 2) << 4) + kh + ((j & 3) << 1);
    v2h p = *(const v2h*)(row + k);
    a[2 * j]     = p.x;
    a[2 * j + 1] = p.y;
  }
  return a;
}

// B fragment from a transposed source Mp[n][k]: two contiguous 16B chunks/lane.
__device__ __forceinline__ v16h load_frag_bt(const _Float16* __restrict__ Mp,
                                             int ldm, int lane) {
  int n  = lane & 15;
  int kb = (lane >> 4) << 3;          // 0 or 8
  const _Float16* row = Mp + (size_t)n * ldm + kb;
  v16h b;
#pragma unroll
  for (int j = 0; j < 8; ++j) {
    v2h p = *(const v2h*)(row + ((j >> 2) << 4) + ((j & 3) << 1));
    b[2 * j]     = p.x;
    b[2 * j + 1] = p.y;
  }
  return b;
}

__device__ __forceinline__ v8f wmma_f16(v16h a, v16h b, v8f c) {
  return __builtin_amdgcn_wmma_f32_16x16x32_f16(false, a, false, b,
                                                (short)0, c, false, false);
}

// C/D layout: m = (lane>>4)*8 + r, n = lane&15.
__device__ __forceinline__ void store_c_f16(v8f c, _Float16* __restrict__ C,
                                            size_t ldc, int lane) {
  int m0 = (lane >> 4) << 3;
  int n  = lane & 15;
#pragma unroll
  for (int r = 0; r < 8; ++r)
    C[(size_t)(m0 + r) * ldc + n] = (_Float16)c[r];
}

// Transposed store: CT[n][m]; each lane writes 8 contiguous f16 (merges wide).
__device__ __forceinline__ void store_ct_f16(v8f c, _Float16* __restrict__ CT,
                                             size_t ldct, int lane) {
  int m0 = (lane >> 4) << 3;
  int n  = lane & 15;
  _Float16* p = CT + (size_t)n * ldct + m0;
#pragma unroll
  for (int r = 0; r < 8; ++r)
    p[r] = (_Float16)c[r];
}

// ---- Tensor Data Mover (CDNA5 TDM) -----------------------------------------
#if __has_builtin(__builtin_amdgcn_tensor_load_to_lds) && \
    __has_builtin(__builtin_amdgcn_s_wait_tensorcnt)
#define USE_TDM 1
typedef __attribute__((ext_vector_type(4))) unsigned int v4u;
typedef __attribute__((ext_vector_type(4))) int          v4i;
typedef __attribute__((ext_vector_type(8))) int          v8i;

// 2D tile load: `rows` rows of `rowlen` f16 elements (row stride = stride
// elements) from global -> contiguous LDS at byte offset lds_off.
// D# packing per CDNA5 ISA ch.8 (group0: count/lds/global/type=2;
// group1: data_size=1 (2B), tensor dims, tile dims, dim0 stride).
__device__ __forceinline__ void tdm_load_2d_to_lds(const void* gsrc,
                                                   unsigned lds_off,
                                                   unsigned rows,
                                                   unsigned rowlen,
                                                   unsigned tensor_rows,
                                                   unsigned stride) {
  unsigned long long ga = (unsigned long long)(uintptr_t)gsrc;
  v4u g0;
  g0.x = 1u;                                            // count = 1 (valid)
  g0.y = lds_off;                                       // lds_addr (bytes)
  g0.z = (unsigned)(ga & 0xffffffffu);                  // global_addr[31:0]
  g0.w = (unsigned)((ga >> 32) & 0x01ffffffu) | (2u << 30); // addr[56:32]|type=2
  v8i g1;
  g1[0] = (int)(1u << 16);                              // data_size = 1 (2B)
  g1[1] = (int)((rowlen & 0xffffu) << 16);              // tensor_dim0[15:0]
  g1[2] = (int)((rowlen >> 16) & 0xffffu) |
          (int)((tensor_rows & 0xffffu) << 16);         // dim0 hi | dim1 lo
  g1[3] = (int)((tensor_rows >> 16) & 0xffffu) |
          (int)((rowlen & 0xffffu) << 16);              // dim1 hi | tile_dim0
  g1[4] = (int)(rows & 0xffffu);                        // tile_dim1 (dim2 = 0)
  g1[5] = (int)stride;                                  // tensor_dim0_stride lo
  g1[6] = 0;
  g1[7] = 0;
  v4i z4 = {0, 0, 0, 0};
#if __clang_major__ >= 23
  v8i z8 = {0, 0, 0, 0, 0, 0, 0, 0};
  __builtin_amdgcn_tensor_load_to_lds(g0, g1, z4, z4, z8, 0);
#else
  __builtin_amdgcn_tensor_load_to_lds(g0, g1, z4, z4, 0);
#endif
}
#endif

// ---- small utility kernels -------------------------------------------------

__global__ void k_zero4(float* p) {
  if (threadIdx.x < 4) p[threadIdx.x] = 0.f;
}

// f32 (R x C, row-major) -> f16 transposed (C x R, row-major)
__global__ void k_f32_to_f16_T(const float* __restrict__ s,
                               _Float16* __restrict__ d, int R, int C) {
  int i = blockIdx.x * 256 + threadIdx.x;
  if (i < R * C) {
    int r = i / C, c = i - r * C;
    d[(size_t)c * R + r] = (_Float16)s[i];
  }
}

// ---- LayerNorm + temporal LIF (threshold 1.0) ------------------------------
__global__ void k_ln_lif(const float* __restrict__ src,
                         const float* __restrict__ g,
                         const float* __restrict__ bb,
                         _Float16* __restrict__ dst,
                         float* acc_slot) {
  __shared__ float red[256];
  int bn  = blockIdx.x;                 // b*256 + n
  int b   = bn >> 8;
  int n   = bn & 255;
  int tid = threadIdx.x;
  int d0  = tid * 2;
  float g0 = g[d0], g1 = g[d0 + 1], bb0 = bb[d0], bb1 = bb[d0 + 1];
  float mem0 = 0.f, mem1 = 0.f, cnt = 0.f;
  for (int t = 0; t < T_; ++t) {
    const float* row = src + ((size_t)(b * T_ + t) * N_ + n) * D_;
    float x0 = row[d0], x1 = row[d0 + 1];
    red[tid] = x0 + x1;
    __syncthreads();
    for (int o = 128; o; o >>= 1) { if (tid < o) red[tid] += red[tid + o]; __syncthreads(); }
    float mean = red[0] * (1.f / (float)D_);
    __syncthreads();
    float dx0 = x0 - mean, dx1 = x1 - mean;
    red[tid] = dx0 * dx0 + dx1 * dx1;
    __syncthreads();
    for (int o = 128; o; o >>= 1) { if (tid < o) red[tid] += red[tid + o]; __syncthreads(); }
    float var = red[0] * (1.f / (float)D_);
    __syncthreads();
    float inv = rsqrtf(var + 1e-5f);
    float h0 = dx0 * inv * g0 + bb0;
    float h1 = dx1 * inv * g1 + bb1;
    mem0 = BETA * mem0 + h0;
    mem1 = BETA * mem1 + h1;
    float s0 = (mem0 >= 1.f) ? 1.f : 0.f;
    float s1 = (mem1 >= 1.f) ? 1.f : 0.f;
    mem0 -= s0;
    mem1 -= s1;
    _Float16* orow = dst + ((size_t)(b * T_ + t) * N_ + n) * D_;
    orow[d0]     = (_Float16)s0;
    orow[d0 + 1] = (_Float16)s1;
    cnt += s0 + s1;
  }
  red[tid] = cnt;
  __syncthreads();
  for (int o = 128; o; o >>= 1) { if (tid < o) red[tid] += red[tid + o]; __syncthreads(); }
  if (tid == 0) atomicAdd(acc_slot, red[0]);
}

// ---- f16 GEMM, C row-major: C[M,N] = A[M,K] * BT[N,K]^T --------------------
// one wave per 32x64 C block: grid = (N/64, M/32), blockDim = 32.
__global__ void k_gemm_f16(const _Float16* __restrict__ A,
                           const _Float16* __restrict__ BT,
                           _Float16* __restrict__ C, int Nn, int K) {
  int lane = threadIdx.x & 31;
  int nb = blockIdx.x * 64;
  int mb = blockIdx.y * 32;
  const _Float16* A0 = A + (size_t)mb * K;
  const _Float16* A1 = A0 + (size_t)16 * K;
  const _Float16* Bb = BT + (size_t)nb * K;
  v8f c00 = {}, c01 = {}, c02 = {}, c03 = {};
  v8f c10 = {}, c11 = {}, c12 = {}, c13 = {};
  for (int kk = 0; kk < K; kk += 32) {
    if (kk + 32 < K) {
      __builtin_prefetch(A0 + (size_t)(lane & 15) * K + kk + 32, 0, 1);
      __builtin_prefetch(Bb + (size_t)(lane & 15) * K + kk + 32, 0, 1);
    }
    v16h a0 = load_frag_a(A0 + kk, K, lane);
    v16h a1 = load_frag_a(A1 + kk, K, lane);
    v16h b0 = load_frag_bt(Bb + kk, K, lane);
    v16h b1 = load_frag_bt(Bb + (size_t)16 * K + kk, K, lane);
    v16h b2 = load_frag_bt(Bb + (size_t)32 * K + kk, K, lane);
    v16h b3 = load_frag_bt(Bb + (size_t)48 * K + kk, K, lane);
    c00 = wmma_f16(a0, b0, c00); c01 = wmma_f16(a0, b1, c01);
    c02 = wmma_f16(a0, b2, c02); c03 = wmma_f16(a0, b3, c03);
    c10 = wmma_f16(a1, b0, c10); c11 = wmma_f16(a1, b1, c11);
    c12 = wmma_f16(a1, b2, c12); c13 = wmma_f16(a1, b3, c13);
  }
  _Float16* Cb = C + (size_t)mb * Nn + nb;
  store_c_f16(c00, Cb,      Nn, lane);
  store_c_f16(c01, Cb + 16, Nn, lane);
  store_c_f16(c02, Cb + 32, Nn, lane);
  store_c_f16(c03, Cb + 48, Nn, lane);
  _Float16* Cb1 = Cb + (size_t)16 * Nn;
  store_c_f16(c10, Cb1,      Nn, lane);
  store_c_f16(c11, Cb1 + 16, Nn, lane);
  store_c_f16(c12, Cb1 + 32, Nn, lane);
  store_c_f16(c13, Cb1 + 48, Nn, lane);
}

// ---- f16 GEMM, C stored TRANSPOSED: CT[N][M] (used for V) ------------------
__global__ void k_gemm_f16_t(const _Float16* __restrict__ A,
                             const _Float16* __restrict__ BT,
                             _Float16* __restrict__ CT, int Nn, int K, int Mtot) {
  int lane = threadIdx.x & 31;
  int nb = blockIdx.x * 64;
  int mb = blockIdx.y * 32;
  const _Float16* A0 = A + (size_t)mb * K;
  const _Float16* A1 = A0 + (size_t)16 * K;
  const _Float16* Bb = BT + (size_t)nb * K;
  v8f c00 = {}, c01 = {}, c02 = {}, c03 = {};
  v8f c10 = {}, c11 = {}, c12 = {}, c13 = {};
  for (int kk = 0; kk < K; kk += 32) {
    if (kk + 32 < K) {
      __builtin_prefetch(A0 + (size_t)(lane & 15) * K + kk + 32, 0, 1);
      __builtin_prefetch(Bb + (size_t)(lane & 15) * K + kk + 32, 0, 1);
    }
    v16h a0 = load_frag_a(A0 + kk, K, lane);
    v16h a1 = load_frag_a(A1 + kk, K, lane);
    v16h b0 = load_frag_bt(Bb + kk, K, lane);
    v16h b1 = load_frag_bt(Bb + (size_t)16 * K + kk, K, lane);
    v16h b2 = load_frag_bt(Bb + (size_t)32 * K + kk, K, lane);
    v16h b3 = load_frag_bt(Bb + (size_t)48 * K + kk, K, lane);
    c00 = wmma_f16(a0, b0, c00); c01 = wmma_f16(a0, b1, c01);
    c02 = wmma_f16(a0, b2, c02); c03 = wmma_f16(a0, b3, c03);
    c10 = wmma_f16(a1, b0, c10); c11 = wmma_f16(a1, b1, c11);
    c12 = wmma_f16(a1, b2, c12); c13 = wmma_f16(a1, b3, c13);
  }
  _Float16* Cb = CT + (size_t)nb * Mtot + mb;
  store_ct_f16(c00, Cb,                     Mtot, lane);
  store_ct_f16(c01, Cb + (size_t)16 * Mtot, Mtot, lane);
  store_ct_f16(c02, Cb + (size_t)32 * Mtot, Mtot, lane);
  store_ct_f16(c03, Cb + (size_t)48 * Mtot, Mtot, lane);
  store_ct_f16(c10, Cb + 16,                     Mtot, lane);
  store_ct_f16(c11, Cb + 16 + (size_t)16 * Mtot, Mtot, lane);
  store_ct_f16(c12, Cb + 16 + (size_t)32 * Mtot, Mtot, lane);
  store_ct_f16(c13, Cb + 16 + (size_t)48 * Mtot, Mtot, lane);
}

// ---- residual GEMM: C[M,N](f32) = X + A[M,K] * BT[N,K]^T -------------------
__global__ void k_gemm_add(const _Float16* __restrict__ A,
                           const _Float16* __restrict__ BT,
                           const float* __restrict__ X,
                           float* __restrict__ C, int Nn, int K) {
  int lane = threadIdx.x & 31;
  int nb = blockIdx.x * 64;
  int mb = blockIdx.y * 32;
  const _Float16* A0 = A + (size_t)mb * K;
  const _Float16* A1 = A0 + (size_t)16 * K;
  const _Float16* Bb = BT + (size_t)nb * K;
  v8f c00 = {}, c01 = {}, c02 = {}, c03 = {};
  v8f c10 = {}, c11 = {}, c12 = {}, c13 = {};
  for (int kk = 0; kk < K; kk += 32) {
    if (kk + 32 < K) {
      __builtin_prefetch(A0 + (size_t)(lane & 15) * K + kk + 32, 0, 1);
      __builtin_prefetch(Bb + (size_t)(lane & 15) * K + kk + 32, 0, 1);
    }
    v16h a0 = load_frag_a(A0 + kk, K, lane);
    v16h a1 = load_frag_a(A1 + kk, K, lane);
    v16h b0 = load_frag_bt(Bb + kk, K, lane);
    v16h b1 = load_frag_bt(Bb + (size_t)16 * K + kk, K, lane);
    v16h b2 = load_frag_bt(Bb + (size_t)32 * K + kk, K, lane);
    v16h b3 = load_frag_bt(Bb + (size_t)48 * K + kk, K, lane);
    c00 = wmma_f16(a0, b0, c00); c01 = wmma_f16(a0, b1, c01);
    c02 = wmma_f16(a0, b2, c02); c03 = wmma_f16(a0, b3, c03);
    c10 = wmma_f16(a1, b0, c10); c11 = wmma_f16(a1, b1, c11);
    c12 = wmma_f16(a1, b2, c12); c13 = wmma_f16(a1, b3, c13);
  }
  int n = nb + (lane & 15);
#pragma unroll
  for (int half = 0; half < 2; ++half) {
    int m0 = mb + half * 16 + ((lane >> 4) << 3);
#pragma unroll
    for (int r = 0; r < 8; ++r) {
      size_t row = (size_t)(m0 + r) * Nn;
      float a = half ? c10[r] : c00[r];
      float b = half ? c11[r] : c01[r];
      float c = half ? c12[r] : c02[r];
      float d = half ? c13[r] : c03[r];
      C[row + n]      = X[row + n]      + a;
      C[row + n + 16] = X[row + n + 16] + b;
      C[row + n + 32] = X[row + n + 32] + c;
      C[row + n + 48] = X[row + n + 48] + d;
    }
  }
}

// ---- fused spiking attention ----------------------------------------------
__global__ void k_attn(const _Float16* __restrict__ q,
                       const _Float16* __restrict__ k,
                       const _Float16* __restrict__ vT,   // [D_][M_]
                       _Float16* __restrict__ attn_out,
                       float* acc_slot) {
  __shared__ float                  mem_s[16 * N_];       // 16 KB
  __shared__ __align__(16) _Float16 sp_s[16 * N_];        // 8 KB
  __shared__ float                  red[4];
  int lane  = threadIdx.x & 31;
  int wave  = threadIdx.x >> 5;          // 0..3
  int ntile = blockIdx.x & 15;
  int h     = (blockIdx.x >> 4) & 7;
  int b     = blockIdx.x >> 7;

  for (int i = threadIdx.x; i < 16 * N_; i += 128) mem_s[i] = 0.f;
  __syncthreads();

  float cnt = 0.f;
  for (int t = 0; t < T_; ++t) {
    size_t rowbase = (size_t)(b * T_ + t) * N_;
    const _Float16* qb = q + (rowbase + ntile * 16) * D_ + h * DH_;

    // ---- scores (16 x 256) + LIF: waves split the 16 key tiles ----
    for (int mt = wave; mt < 16; mt += 4) {
      const _Float16* kb = k + (rowbase + mt * 16) * D_ + h * DH_;
      v8f c = {};
#pragma unroll
      for (int kk = 0; kk < DH_; kk += 32) {
        v16h a  = load_frag_a(qb + kk, D_, lane);
        v16h bf = load_frag_bt(kb + kk, D_, lane);   // k^T: contiguous pairs
        c = wmma_f16(a, bf, c);
      }
      int lr0 = (lane >> 4) << 3;
      int col = mt * 16 + (lane & 15);
#pragma unroll
      for (int r = 0; r < 8; ++r) {
        int idx = (lr0 + r) * N_ + col;
        float m2 = BETA * mem_s[idx] + c[r] * 0.125f;  // scale = Dh^-0.5
        float s  = (m2 >= 0.5f) ? 1.f : 0.f;
        mem_s[idx] = m2 - 0.5f * s;
        sp_s[idx]  = (_Float16)s;
        cnt += s;
      }
    }
    __syncthreads();

    // ---- out (16 x 64) = spikes(16 x 256) @ v_t(256 x 64): waves split Dh --
    {
      int dt = wave;                                   // 16-col subtile of Dh
      const _Float16* vb = vT + (size_t)(h * DH_ + dt * 16) * M_ + rowbase;
      v8f c = {};
#pragma unroll
      for (int mc = 0; mc < 8; ++mc) {                 // K chunks of 32 over 256
        v16h a  = load_frag_a(sp_s + mc * 32, N_, lane);        // from LDS
        v16h bf = load_frag_bt(vb + mc * 32, M_, lane);         // contiguous
        c = wmma_f16(a, bf, c);
      }
      int lr0 = (lane >> 4) << 3;
      int n   = lane & 15;
      _Float16* ob = attn_out + (rowbase + ntile * 16) * D_ + h * DH_ + dt * 16;
#pragma unroll
      for (int r = 0; r < 8; ++r)
        ob[(size_t)(lr0 + r) * D_ + n] = (_Float16)c[r];
    }
    __syncthreads();
  }

  for (int o = 16; o; o >>= 1) cnt += __shfl_xor(cnt, o, 32);
  if (lane == 0) red[wave] = cnt;
  __syncthreads();
  if (threadIdx.x == 0)
    atomicAdd(acc_slot, red[0] + red[1] + red[2] + red[3]);
}

// ---- fused MLP-in GEMM + temporal LIF (TDM-staged weights) -----------------
// One wave per 16x64 hid strip. The 64x512 f16 weight strip (64 KB) is moved
// to LDS once by the Tensor Data Mover and reused across the T=4 time steps;
// the LIF membrane lives in the f32 accumulator registers, so the 67 MB hid
// tensor never hits memory. grid = (HID/64, B*N/16), block = 32.
__global__ void k_gemm_lif(const _Float16* __restrict__ A,    // s2 (B,T,N,D)
                           const _Float16* __restrict__ BT,   // w1^T (HID,D)
                           _Float16* __restrict__ hs,         // (B,T,N,HID)
                           float* acc_slot) {
  __shared__ __align__(16) _Float16 bsh[64 * D_];             // 64 KB strip
  int lane = threadIdx.x & 31;
  int colb = blockIdx.x * 64;          // over HID
  int r0   = blockIdx.y * 16;          // over b*256+n (tiles never straddle b)
  int b    = r0 >> 8;
  int nb   = r0 & 255;
  const _Float16* Bg = BT + (size_t)colb * D_;

#ifdef USE_TDM
  tdm_load_2d_to_lds(Bg, (unsigned)(uintptr_t)(void*)bsh,
                     /*rows=*/64, /*rowlen=*/D_,
                     /*tensor_rows=*/HID_, /*stride=*/D_);
  __builtin_amdgcn_s_wait_tensorcnt(0);
#else
  {
    const unsigned long long* srcp = (const unsigned long long*)Bg;
    unsigned long long* dstp = (unsigned long long*)bsh;
    for (int i = lane; i < 64 * D_ / 4; i += 32) dstp[i] = srcp[i];
    __syncthreads();
  }
#endif

  v8f m0v = {}, m1v = {}, m2v = {}, m3v = {};
  float cnt = 0.f;
  for (int t = 0; t < T_; ++t) {
    size_t row0 = (size_t)(b * T_ + t) * N_ + nb;
    const _Float16* Ab = A + row0 * D_;
    v8f c0 = {}, c1 = {}, c2 = {}, c3 = {};
    for (int kk = 0; kk < D_; kk += 32) {
      if (kk + 32 < D_)
        __builtin_prefetch(Ab + (size_t)(lane & 15) * D_ + kk + 32, 0, 1);
      v16h a  = load_frag_a(Ab + kk, D_, lane);
      v16h b0 = load_frag_bt(bsh + kk, D_, lane);               // from LDS
      v16h b1 = load_frag_bt(bsh + (size_t)16 * D_ + kk, D_, lane);
      v16h b2 = load_frag_bt(bsh + (size_t)32 * D_ + kk, D_, lane);
      v16h b3 = load_frag_bt(bsh + (size_t)48 * D_ + kk, D_, lane);
      c0 = wmma_f16(a, b0, c0);
      c1 = wmma_f16(a, b1, c1);
      c2 = wmma_f16(a, b2, c2);
      c3 = wmma_f16(a, b3, c3);
    }
    int lr0 = (lane >> 4) << 3;
    int n   = lane & 15;
#pragma unroll
    for (int r = 0; r < 8; ++r) {
      _Float16* orow = hs + (row0 + lr0 + r) * HID_ + colb + n;
      float m2, s;
      m2 = BETA * m0v[r] + c0[r]; s = (m2 >= 1.f) ? 1.f : 0.f; m0v[r] = m2 - s; orow[0]  = (_Float16)s; cnt += s;
      m2 = BETA * m1v[r] + c1[r]; s = (m2 >= 1.f) ? 1.f : 0.f; m1v[r] = m2 - s; orow[16] = (_Float16)s; cnt += s;
      m2 = BETA * m2v[r] + c2[r]; s = (m2 >= 1.f) ? 1.f : 0.f; m2v[r] = m2 - s; orow[32] = (_Float16)s; cnt += s;
      m2 = BETA * m3v[r] + c3[r]; s = (m2 >= 1.f) ? 1.f : 0.f; m3v[r] = m2 - s; orow[48] = (_Float16)s; cnt += s;
    }
  }
  for (int o = 16; o; o >>= 1) cnt += __shfl_xor(cnt, o, 32);
  if (lane == 0) atomicAdd(acc_slot, cnt);
}

__global__ void k_final(const float* __restrict__ acc, float* __restrict__ tail) {
  if (threadIdx.x == 0) {
    tail[0] = acc[0] * (1.f / 4194304.f);    // r_lif1: B*T*N*D
    tail[1] = acc[1] * (1.f / 16777216.f);   // r_attn: B*T*H*N*N
    tail[2] = acc[2] * (1.f / 4194304.f);    // r_lif2
    tail[3] = acc[3] * (1.f / 16777216.f);   // r_ffn : B*T*N*HID
  }
}

// ---------------------------------------------------------------------------

extern "C" void kernel_launch(void* const* d_in, const int* in_sizes, int n_in,
                              void* d_out, int out_size, void* d_ws, size_t ws_size,
                              hipStream_t stream) {
  const float* x  = (const float*)d_in[0];
  const float* g1 = (const float*)d_in[1];
  const float* b1 = (const float*)d_in[2];
  const float* wq = (const float*)d_in[3];
  const float* wk = (const float*)d_in[4];
  const float* wv = (const float*)d_in[5];
  const float* wo = (const float*)d_in[6];
  const float* g2 = (const float*)d_in[7];
  const float* b2 = (const float*)d_in[8];
  const float* w1 = (const float*)d_in[9];
  const float* w2 = (const float*)d_in[10];
  float* out = (float*)d_out;

  // workspace layout (all weight copies transposed, f16)
  char* ws = (char*)d_ws;
  float*    acc  = (float*)ws;                       // 4 accumulators
  _Float16* wqt  = (_Float16*)(ws + 256);            // (D,D)^T
  _Float16* wkt  = wqt + 262144;
  _Float16* wvt  = wkt + 262144;
  _Float16* wot  = wvt + 262144;
  _Float16* w1t  = wot + 262144;                     // (HID,D)
  _Float16* w2t  = w1t + 1048576;                    // (D,HID)
  _Float16* sbuf = w2t + 1048576;                    // spikes s1/s2 (M_,D)
  _Float16* qh   = sbuf + 4194304;                   // (M_,D)
  _Float16* kh   = qh + 4194304;                     // (M_,D)
  _Float16* vth  = kh + 4194304;                     // V^T (D,M_)
  _Float16* aoh  = vth + 4194304;                    // attention out (M_,D)
  _Float16* hsh  = aoh + 4194304;                    // hs (M_,HID)

  k_zero4<<<1, 32, 0, stream>>>(acc);

  // weights -> f16 transposed
  k_f32_to_f16_T<<<(262144 + 255) / 256, 256, 0, stream>>>(wq, wqt, D_, D_);
  k_f32_to_f16_T<<<(262144 + 255) / 256, 256, 0, stream>>>(wk, wkt, D_, D_);
  k_f32_to_f16_T<<<(262144 + 255) / 256, 256, 0, stream>>>(wv, wvt, D_, D_);
  k_f32_to_f16_T<<<(262144 + 255) / 256, 256, 0, stream>>>(wo, wot, D_, D_);
  k_f32_to_f16_T<<<(1048576 + 255) / 256, 256, 0, stream>>>(w1, w1t, D_, HID_);
  k_f32_to_f16_T<<<(1048576 + 255) / 256, 256, 0, stream>>>(w2, w2t, HID_, D_);

  // ---- attention branch ----
  k_ln_lif<<<B_ * N_, 256, 0, stream>>>(x, g1, b1, sbuf, acc + 0);
  k_gemm_f16  <<<dim3(D_ / 64, M_ / 32), 32, 0, stream>>>(sbuf, wqt, qh, D_, D_);
  k_gemm_f16  <<<dim3(D_ / 64, M_ / 32), 32, 0, stream>>>(sbuf, wkt, kh, D_, D_);
  k_gemm_f16_t<<<dim3(D_ / 64, M_ / 32), 32, 0, stream>>>(sbuf, wvt, vth, D_, D_, M_);
  k_attn<<<B_ * H_ * (N_ / 16), 128, 0, stream>>>(qh, kh, vth, aoh, acc + 1);
  k_gemm_add<<<dim3(D_ / 64, M_ / 32), 32, 0, stream>>>(aoh, wot, x, out, D_, D_);

  // ---- MLP branch ----
  k_ln_lif<<<B_ * N_, 256, 0, stream>>>(out, g2, b2, sbuf, acc + 2);
  k_gemm_lif<<<dim3(HID_ / 64, (B_ * N_) / 16), 32, 0, stream>>>(sbuf, w1t, hsh, acc + 3);
  k_gemm_add<<<dim3(D_ / 64, M_ / 32), 32, 0, stream>>>(hsh, w2t, out, out, D_, HID_);

  k_final<<<1, 32, 0, stream>>>(acc, out + 4194304);

  (void)in_sizes; (void)n_in; (void)out_size; (void)ws_size;
}
